// MatchAttn_48155173323066
// MI455X (gfx1250) — compile-verified, matrix operand order checked by reading, not run on hardware
//
#include <hip/hip_runtime.h>
#include <hip/hip_bf16.h>

// ---------------------------------------------------------------------------
// MatchAttn for MI455X (gfx1250):
//   - all GEMMs via v_wmma_f32_16x16x32_bf16 (f32 accumulate)
//   - global->LDS tile staging via the Tensor Data Mover (tensor_load_to_lds),
//     double-buffered, synchronized with s_wait_tensorcnt + barriers
// ---------------------------------------------------------------------------

typedef __attribute__((ext_vector_type(16))) __bf16        bf16x16;
typedef __attribute__((ext_vector_type(8)))  float         f32x8;
typedef __attribute__((ext_vector_type(4)))  unsigned int  u32x4;
typedef __attribute__((ext_vector_type(8)))  int           i32x8;
typedef __attribute__((ext_vector_type(4)))  int           i32x4;

union AFrag { bf16x16 v; uint4 q[2]; unsigned short u[16]; };
union CFrag { f32x8 v; float f[8]; };

__device__ __forceinline__ unsigned short f32_to_bf16_rne(float f) {
    unsigned int u = __float_as_uint(f);
    unsigned int r = u + 0x7FFFu + ((u >> 16) & 1u);
    return (unsigned short)(r >> 16);
}

// ---------------------------------------------------------------------------
// TDM: DMA one [rows x 32] bf16 tile (row stride = ld elements) from global
// memory into LDS at byte offset lds_off (rows stored contiguously, 64B each).
// D# per CDNA5 ISA 8.3/8.4: group0 = {count=1, lds_addr, global_addr, type=2},
// group1 = {data_size=2B, tensor_dim0=32, tensor_dim1=rows, tile_dim0=32,
//           tile_dim1=rows, tensor_dim0_stride=ld}.  2D tensor -> groups 2/3
// zero.  Issued by one wave; tracked with TENSORcnt.
// ---------------------------------------------------------------------------
__device__ __forceinline__ void tdm_load_tile(unsigned lds_off,
                                              const unsigned short* gptr,
                                              int rows, int ld) {
    unsigned long long ga = (unsigned long long)gptr;
    u32x4 g0;
    g0[0] = 1u;                                            // count=1, valid
    g0[1] = lds_off;                                       // LDS byte address
    g0[2] = (unsigned)ga;                                  // global addr [31:0]
    g0[3] = (unsigned)((ga >> 32) & 0x1FFFFFFu) | (2u << 30); // [56:32] | type=2
    i32x8 g1;
    g1[0] = 0x00010000;        // workgroup_mask=0, data_size=1 (2 bytes)
    g1[1] = (32  << 16);       // tensor_dim0 = 32  (bits 63:48)
    g1[2] = (rows << 16);      // tensor_dim1 = rows (bits 95:80)
    g1[3] = (32  << 16);       // tile_dim0 = 32    (bits 127:112)
    g1[4] = rows;              // tile_dim1 = rows  (bits 143:128)
    g1[5] = ld;                // tensor_dim0_stride (bits 191:160)
    g1[6] = 0;
    g1[7] = 0;
    i32x4 z4 = (i32x4){0, 0, 0, 0};
#if defined(__clang_major__) && (__clang_major__ >= 23)
    i32x8 z8 = (i32x8){0, 0, 0, 0, 0, 0, 0, 0};
    __builtin_amdgcn_tensor_load_to_lds(g0, g1, z4, z4, z8, 0);
#else
    __builtin_amdgcn_tensor_load_to_lds(g0, g1, z4, z4, 0);
#endif
}

// Load one 16x32 bf16 fragment from an LDS tile row block (row stride 32).
// A-operand lane layout (ISA 7.12.2):
//   lanes 0-15 : row = lane,    K-chunks [0..7]  and [16..23]
//   lanes 16-31: row = lane-16, K-chunks [8..15] and [24..31]
__device__ __forceinline__ bf16x16 lds_frag16x32(const unsigned short* p0) {
    const int lane = threadIdx.x & 31;
    const int half = lane >> 4;
    const int r    = lane & 15;
    const unsigned short* p = p0 + r * 32 + half * 8;
    AFrag f;
    f.q[0] = *(const uint4*)(p);
    f.q[1] = *(const uint4*)(p + 16);
    return f.v;
}

// ---------------------------------------------------------------------------
// bf16 WMMA GEMM: C[b] = act(A[b] (MxK) * Bt[b]^T (KxN) + bias)
// Bt stored row-major as (N x K).  Block: 256 threads = 8 waves,
// block tile 128(M) x 128(N), K-step 32.  Wave tile 64x32 (4x2 frags).
// TDM double-buffers A(128x32) and B(128x32) tiles in LDS.
// ---------------------------------------------------------------------------
template<bool HAS_BIAS, bool RELU, bool OUT_BF16>
__global__ __launch_bounds__(256)
void gemm_bf16_wmma(const unsigned short* __restrict__ A,
                    const unsigned short* __restrict__ Bt,
                    const float* __restrict__ bias,
                    float* __restrict__ Cf,
                    unsigned short* __restrict__ Cbf,
                    int M, int N, int K,
                    long long sA, long long sB, long long sC) {
    __shared__ unsigned short shA[2][128][32];
    __shared__ unsigned short shB[2][128][32];

    const int b = blockIdx.z;
    A  += (size_t)b * sA;
    Bt += (size_t)b * sB;

    const int wave = threadIdx.x >> 5;       // 8 waves (wave32)
    const int wm   = wave & 1;               // 2 waves along M
    const int wn   = wave >> 1;              // 4 waves along N
    const int mloc = wm * 64;                // wave row origin in block tile
    const int nloc = wn * 32;                // wave col origin in block tile
    const int mblk = blockIdx.y * 128;
    const int nblk = blockIdx.x * 128;
    const bool issuer = (wave == 0);

    f32x8 acc[4][2];
#pragma unroll
    for (int i = 0; i < 4; ++i)
#pragma unroll
        for (int j = 0; j < 2; ++j)
            acc[i][j] = (f32x8){0.f, 0.f, 0.f, 0.f, 0.f, 0.f, 0.f, 0.f};

    const unsigned ldsA0 = (unsigned)(size_t)&shA[0][0][0];
    const unsigned ldsA1 = (unsigned)(size_t)&shA[1][0][0];
    const unsigned ldsB0 = (unsigned)(size_t)&shB[0][0][0];
    const unsigned ldsB1 = (unsigned)(size_t)&shB[1][0][0];

    // prologue: DMA first K tile into buffer 0
    if (issuer) {
        tdm_load_tile(ldsA0, A  + (size_t)mblk * K, 128, K);
        tdm_load_tile(ldsB0, Bt + (size_t)nblk * K, 128, K);
    }

    int buf = 0;
    for (int k = 0; k < K; k += 32) {
        if (issuer) __builtin_amdgcn_s_wait_tensorcnt(0);
        __syncthreads();   // current buffer ready; previous compute finished

        if (issuer && (k + 32 < K)) {  // DMA next K tile into the back buffer
            const unsigned la = buf ? ldsA0 : ldsA1;
            const unsigned lb = buf ? ldsB0 : ldsB1;
            tdm_load_tile(la, A  + (size_t)mblk * K + (k + 32), 128, K);
            tdm_load_tile(lb, Bt + (size_t)nblk * K + (k + 32), 128, K);
        }

        bf16x16 af[4], bfr[2];
#pragma unroll
        for (int i = 0; i < 4; ++i)
            af[i] = lds_frag16x32(&shA[buf][mloc + 16 * i][0]);
#pragma unroll
        for (int j = 0; j < 2; ++j)
            bfr[j] = lds_frag16x32(&shB[buf][nloc + 16 * j][0]);

#pragma unroll
        for (int i = 0; i < 4; ++i)
#pragma unroll
            for (int j = 0; j < 2; ++j)
                acc[i][j] = __builtin_amdgcn_wmma_f32_16x16x32_bf16(
                    false, af[i], false, bfr[j], (short)0, acc[i][j],
                    false, false);

        buf ^= 1;
    }

    // C layout (ISA 7.12.2, 32-bit 16x16): VGPR r -> M = r + 8*(lane>=16), N = lane&15
    const int lane = threadIdx.x & 31;
    const int half = lane >> 4;
    const int nlo  = lane & 15;
#pragma unroll
    for (int i = 0; i < 4; ++i) {
#pragma unroll
        for (int j = 0; j < 2; ++j) {
            CFrag c; c.v = acc[i][j];
            const int col = nblk + nloc + j * 16 + nlo;
            const float bv = HAS_BIAS ? bias[col] : 0.f;
#pragma unroll
            for (int r = 0; r < 8; ++r) {
                const int row = mblk + mloc + i * 16 + half * 8 + r;
                float v = c.f[r] + bv;
                if (RELU) v = v > 0.f ? v : 0.f;
                const size_t idx = (size_t)b * sC + (size_t)row * N + col;
                if (OUT_BF16) Cbf[idx] = f32_to_bf16_rne(v);
                else          Cf[idx]  = v;
            }
        }
    }
}

// ---------------------------------------------------------------------------
// f32 -> bf16 elementwise convert (4 elems/thread, vectorized)
// ---------------------------------------------------------------------------
__global__ __launch_bounds__(256)
void convert_f32_to_bf16(const float* __restrict__ in,
                         unsigned short* __restrict__ out, long long n) {
    long long i4 = ((long long)blockIdx.x * blockDim.x + threadIdx.x) * 4;
    if (i4 + 3 < n) {
        float4 f = *(const float4*)(in + i4);
        union { unsigned short u[4]; uint2 v; } o;
        o.u[0] = f32_to_bf16_rne(f.x);
        o.u[1] = f32_to_bf16_rne(f.y);
        o.u[2] = f32_to_bf16_rne(f.z);
        o.u[3] = f32_to_bf16_rne(f.w);
        *(uint2*)(out + i4) = o.v;
    }
}

// ---------------------------------------------------------------------------
// Per-batch transpose + convert: yt[b][d][l] = bf16(y[b][l][d])  (LDS tiled)
// ---------------------------------------------------------------------------
__global__ __launch_bounds__(256)
void transpose_convert_bf16(const float* __restrict__ y,
                            unsigned short* __restrict__ yt, int L, int D) {
    __shared__ float tile[32][33];
    const int b  = blockIdx.z;
    const float* yb = y + (size_t)b * L * D;
    unsigned short* ytb = yt + (size_t)b * L * D;
    const int d0 = blockIdx.x * 32;
    const int l0 = blockIdx.y * 32;
#pragma unroll
    for (int s = 0; s < 32; s += 8)
        tile[threadIdx.y + s][threadIdx.x] =
            yb[(size_t)(l0 + threadIdx.y + s) * D + d0 + threadIdx.x];
    __syncthreads();
#pragma unroll
    for (int s = 0; s < 32; s += 8) {
        const int d = threadIdx.y + s;
        const int l = threadIdx.x;
        ytb[(size_t)(d0 + d) * L + l0 + l] = f32_to_bf16_rne(tile[l][d]);
    }
}

// ---------------------------------------------------------------------------
// Masked row softmax over L2: where mask!=0 -> -inf (reference semantics).
// Writes f32 alpha in place and a bf16 copy for the final WMMA GEMM.
// ---------------------------------------------------------------------------
__global__ __launch_bounds__(256)
void masked_softmax(float* __restrict__ S, const int* __restrict__ mask,
                    unsigned short* __restrict__ Abf, int L1, int L2) {
    __shared__ float red[256];
    const int row = blockIdx.x;
    const int b   = row / L1;
    float* s = S + (size_t)row * L2;
    const int* mk = mask + (size_t)b * L2;
    const int t = threadIdx.x;

    float vals[4];
    float mx = -__builtin_inff();
#pragma unroll
    for (int k = 0; k < 4; ++k) {
        const int j = t + k * 256;
        const float v = (mk[j] != 0) ? -__builtin_inff() : s[j];
        vals[k] = v;
        mx = fmaxf(mx, v);
    }
    red[t] = mx; __syncthreads();
    for (int off = 128; off > 0; off >>= 1) {
        if (t < off) red[t] = fmaxf(red[t], red[t + off]);
        __syncthreads();
    }
    mx = red[0]; __syncthreads();

    float sum = 0.f;
#pragma unroll
    for (int k = 0; k < 4; ++k) { float e = __expf(vals[k] - mx); vals[k] = e; sum += e; }
    red[t] = sum; __syncthreads();
    for (int off = 128; off > 0; off >>= 1) {
        if (t < off) red[t] += red[t + off];
        __syncthreads();
    }
    const float inv = 1.f / red[0];

    unsigned short* ab = Abf + (size_t)row * L2;
#pragma unroll
    for (int k = 0; k < 4; ++k) {
        const int j = t + k * 256;
        const float a = vals[k] * inv;
        s[j]  = a;
        ab[j] = f32_to_bf16_rne(a);
    }
}

// ---------------------------------------------------------------------------
extern "C" void kernel_launch(void* const* d_in, const int* in_sizes, int n_in,
                              void* d_out, int out_size, void* d_ws, size_t ws_size,
                              hipStream_t stream) {
    constexpr int B = 16, L1 = 1024, L2 = 1024, D = 1024;
    constexpr long long XE = (long long)B * L1 * D;  // 16M elems
    constexpr long long YE = (long long)B * L2 * D;
    constexpr long long WE = (long long)D * D;

    const float* x     = (const float*)d_in[0];
    const float* y     = (const float*)d_in[1];
    const int*   ymask = (const int*)  d_in[2];
    const float* Wx    = (const float*)d_in[3];
    const float* bx    = (const float*)d_in[4];
    const float* Wy    = (const float*)d_in[5];
    const float* by    = (const float*)d_in[6];
    const float* W     = (const float*)d_in[7];

    float* matched = (float*)d_out;        // (B, L1, D) f32
    float* alpha   = (float*)d_out + XE;   // (B, L1, L2) f32 (scores -> softmax in place)

    // Workspace (bytes). Reuse: Xw overwrites Xbf, AlphaBf overwrites Ybf.
    char* ws = (char*)d_ws;
    unsigned short* Xbf = (unsigned short*)(ws);                       // 2*XE
    unsigned short* Ybf = (unsigned short*)(ws + 2 * XE);              // 2*YE
    unsigned short* Ybt = (unsigned short*)(ws + 4 * XE);              // 2*YE (Y^T per batch)
    unsigned short* Wxb = (unsigned short*)(ws + 6 * XE);              // 2*WE
    unsigned short* Wyb = (unsigned short*)(ws + 6 * XE + 2 * WE);
    unsigned short* Wb  = (unsigned short*)(ws + 6 * XE + 4 * WE);
    unsigned short* Xp  = (unsigned short*)(ws + 6 * XE + 6 * WE);     // 2*XE
    unsigned short* Yp  = (unsigned short*)(ws + 8 * XE + 6 * WE);     // 2*YE
    unsigned short* Xw  = Xbf;   // reuse after gemm1 consumed Xbf
    unsigned short* Abf = Ybf;   // reuse after gemm2 consumed Ybf

    const dim3 t256(256);

    // --- converts ---
    convert_f32_to_bf16<<<dim3((unsigned)(XE / 4 / 256)), t256, 0, stream>>>(x, Xbf, XE);
    convert_f32_to_bf16<<<dim3((unsigned)(YE / 4 / 256)), t256, 0, stream>>>(y, Ybf, YE);
    convert_f32_to_bf16<<<dim3((unsigned)(WE / 4 / 256)), t256, 0, stream>>>(Wx, Wxb, WE);
    convert_f32_to_bf16<<<dim3((unsigned)(WE / 4 / 256)), t256, 0, stream>>>(Wy, Wyb, WE);
    convert_f32_to_bf16<<<dim3((unsigned)(WE / 4 / 256)), t256, 0, stream>>>(W,  Wb,  WE);
    transpose_convert_bf16<<<dim3(D / 32, L2 / 32, B), dim3(32, 8), 0, stream>>>(y, Ybt, L2, D);

    // --- projections: x_proj = relu(X*Wx^T + bx), y_proj = relu(Y*Wy^T + by) ---
    gemm_bf16_wmma<true, true, true><<<dim3(D / 128, (B * L1) / 128, 1), t256, 0, stream>>>(
        Xbf, Wxb, bx, nullptr, Xp, B * L1, D, D, 0, 0, 0);
    gemm_bf16_wmma<true, true, true><<<dim3(D / 128, (B * L2) / 128, 1), t256, 0, stream>>>(
        Ybf, Wyb, by, nullptr, Yp, B * L2, D, D, 0, 0, 0);

    // --- x_proj = x_proj * W^T ---
    gemm_bf16_wmma<false, false, true><<<dim3(D / 128, (B * L1) / 128, 1), t256, 0, stream>>>(
        Xp, Wb, nullptr, nullptr, Xw, B * L1, D, D, 0, 0, 0);

    // --- scores[b] = Xw[b] * Yp[b]^T -> f32 into alpha region of d_out ---
    gemm_bf16_wmma<false, false, false><<<dim3(L2 / 128, L1 / 128, B), t256, 0, stream>>>(
        Xw, Yp, nullptr, alpha, nullptr, L1, L2, D,
        (long long)L1 * D, (long long)L2 * D, (long long)L1 * L2);

    // --- mask + softmax (f32 in place, bf16 copy for final GEMM) ---
    masked_softmax<<<dim3(B * L1), t256, 0, stream>>>(alpha, ymask, Abf, L1, L2);

    // --- matched[b] = alpha[b] * Y[b]  (B operand = Ybt rows) ---
    gemm_bf16_wmma<false, false, false><<<dim3(D / 128, L1 / 128, B), t256, 0, stream>>>(
        Abf, Ybt, nullptr, matched, nullptr, L1, D, L2,
        (long long)L1 * L2, (long long)D * L2, (long long)L1 * D);
}
